// EGNNLayer_56324201119978
// MI455X (gfx1250) — matrix operand order repaired and checked by reading, not compiled, not run on hardware
//
#include <hip/hip_runtime.h>
#include <hip/hip_bf16.h>

// ---------------------------------------------------------------------------
// EGNN layer for MI455X (gfx1250): fused edge-MLP / scatter / node-MLP using
// v_wmma_f32_16x16x32_f16 (f16 inputs, f32 accumulate). One wave32 per
// 16-row tile; E/16 = 50000 and N/16 = 3125 exactly (no tail handling).
// ---------------------------------------------------------------------------

typedef __attribute__((ext_vector_type(16))) _Float16 v16h;
typedef __attribute__((ext_vector_type(8)))  _Float16 v8h;
typedef __attribute__((ext_vector_type(8)))  float    v8f;

#define N_NODES 50000
#define N_EDGES 800000
#define HDIM    128

__device__ __forceinline__ float silu_f(float v) { return v / (1.f + __expf(-v)); }

// A-matrix 16x32 f16 fragment (ISA 7.12.2): lane L holds row M=L&15;
// halves [0..7] = K = k0 + 8*(L>>4) + i ; halves [8..15] = K = k0 + 16 + 8*(L>>4) + i
// p must point at element (M, k0 + 8*half).
__device__ __forceinline__ v16h frag_a(const _Float16* p) {
  v8h lo = *reinterpret_cast<const v8h*>(p);
  v8h hi = *reinterpret_cast<const v8h*>(p + 16);
  v16h a;
#pragma unroll
  for (int i = 0; i < 8; ++i) { a[i] = lo[i]; a[i + 8] = hi[i]; }
  return a;
}

// B-matrix 32x16 f16 fragment: lane L holds col N=L&15; halves [0..7] =
// K = k0 + 8*(L>>4) + i ; halves [8..15] = K = k0 + 16 + 8*(L>>4) + i (matches A).
// p must point at W[n*Kdim + k0 + 8*half] (W stored row-major (out,K)).
__device__ __forceinline__ v16h frag_b(const _Float16* p) {
  v8h lo = *reinterpret_cast<const v8h*>(p);
  v8h hi = *reinterpret_cast<const v8h*>(p + 16);
  v16h b;
#pragma unroll
  for (int i = 0; i < 8; ++i) { b[i] = lo[i]; b[i + 8] = hi[i]; }
  return b;
}

#define WMMA(a, b, c) __builtin_amdgcn_wmma_f32_16x16x32_f16(false, (a), false, (b), (short)0, (c), false, false)

// ---------------------------------------------------------------------------
// 0) zero scratch accumulators
__global__ void egnn_zero_kernel(float* p, long long n) {
  long long i = (long long)blockIdx.x * 256 + threadIdx.x;
  if (i < n) p[i] = 0.f;
}

// 1) weights fp32 -> fp16, row-major (out, K)
__global__ void egnn_cvt_kernel(const float* __restrict__ We1, const float* __restrict__ We2,
                                const float* __restrict__ Wn1, const float* __restrict__ Wn2,
                                const float* __restrict__ Wc1,
                                _Float16* We1h, _Float16* We2h, _Float16* Wn1h,
                                _Float16* Wn2h, _Float16* Wc1h) {
  int n = blockIdx.x;      // 0..127
  int t = threadIdx.x;     // 0..255
  We1h[n * 256 + t] = (_Float16)We1[n * 257 + t];   // drop col 256 (handled in f32)
  Wn1h[n * 256 + t] = (_Float16)Wn1[n * 256 + t];
  if (t < 128) {
    We2h[n * 128 + t] = (_Float16)We2[n * 128 + t];
    Wn2h[n * 128 + t] = (_Float16)Wn2[n * 128 + t];
    Wc1h[n * 128 + t] = (_Float16)Wc1[n * 128 + t];
  }
}

// ---------------------------------------------------------------------------
// 2) fused edge pipeline: one wave per 16 edges.
__global__ __launch_bounds__(32) void egnn_edge_kernel(
    const float* __restrict__ h, const float* __restrict__ x,
    const long long* __restrict__ ei,
    const float* __restrict__ We1,  // f32, for the dist_sq column (K=256)
    const float* __restrict__ be1, const float* __restrict__ be2,
    const float* __restrict__ bc1, const float* __restrict__ Wc2,
    const _Float16* __restrict__ We1h, const _Float16* __restrict__ We2h,
    const _Float16* __restrict__ Wc1h,
    float* __restrict__ agg, float* __restrict__ cnt, float* __restrict__ coordAgg) {
  __shared__ _Float16 A[16 * 256];    // [edge][h_src(128) | h_dst(128)]
  __shared__ _Float16 M1[16 * 128];   // reused: m1, then m
  __shared__ float distS[16];
  __shared__ float diffS[16 * 3];
  __shared__ int   dstS[16];
  __shared__ float cwS[16];

  const int lane = threadIdx.x;
  const long long eBase = (long long)blockIdx.x * 16;
  const int nl = lane & 15;
  const int hh = lane >> 4;

  // --- edge metadata -------------------------------------------------------
  if (lane < 16) {
    long long e = eBase + lane;
    int s = (int)ei[e];
    int d = (int)ei[(long long)N_EDGES + e];
    dstS[lane] = d;
    float dx = x[s * 3 + 0] - x[d * 3 + 0];
    float dy = x[s * 3 + 1] - x[d * 3 + 1];
    float dz = x[s * 3 + 2] - x[d * 3 + 2];
    diffS[lane * 3 + 0] = dx; diffS[lane * 3 + 1] = dy; diffS[lane * 3 + 2] = dz;
    distS[lane] = dx * dx + dy * dy + dz * dz;
    cwS[lane] = 0.f;
  }
  __syncthreads();

  // --- gather h[src], h[dst] into the f16 A-tile ---------------------------
  for (int r = 0; r < 32; ++r) {
    int e16 = r & 15;
    long long e = eBase + e16;
    int row = (r < 16) ? (int)ei[e] : (int)ei[(long long)N_EDGES + e];
    float4 v = *reinterpret_cast<const float4*>(h + (long long)row * HDIM + lane * 4);
    _Float16* dp = &A[e16 * 256 + (r < 16 ? 0 : 128) + lane * 4];
    dp[0] = (_Float16)v.x; dp[1] = (_Float16)v.y;
    dp[2] = (_Float16)v.z; dp[3] = (_Float16)v.w;
  }
  __syncthreads();

  v8f acc[8];

  // --- GEMM1: m1 = silu([h_s|h_d|dist] @ We1.T + be1), K=256 + rank-1 dist --
#pragma unroll
  for (int nt = 0; nt < 8; ++nt) {
    int n = nt * 16 + nl;
    float b0 = be1[n];
    float wd = We1[n * 257 + 256];
#pragma unroll
    for (int j = 0; j < 8; ++j) acc[nt][j] = b0 + distS[j + 8 * hh] * wd;
  }
#pragma unroll
  for (int kt = 0; kt < 8; ++kt) {
    v16h a = frag_a(&A[nl * 256 + kt * 32 + 8 * hh]);
#pragma unroll
    for (int nt = 0; nt < 8; ++nt) {
      v16h b = frag_b(&We1h[(nt * 16 + nl) * 256 + kt * 32 + 8 * hh]);
      acc[nt] = WMMA(a, b, acc[nt]);
    }
  }
#pragma unroll
  for (int nt = 0; nt < 8; ++nt) {
    int n = nt * 16 + nl;
#pragma unroll
    for (int j = 0; j < 8; ++j)
      M1[(j + 8 * hh) * 128 + n] = (_Float16)silu_f(acc[nt][j]);
  }
  __syncthreads();

  // --- GEMM2: m = silu(m1 @ We2.T + be2), K=128 ----------------------------
#pragma unroll
  for (int nt = 0; nt < 8; ++nt) {
    float b0 = be2[nt * 16 + nl];
#pragma unroll
    for (int j = 0; j < 8; ++j) acc[nt][j] = b0;
  }
#pragma unroll
  for (int kt = 0; kt < 4; ++kt) {
    v16h a = frag_a(&M1[nl * 128 + kt * 32 + 8 * hh]);
#pragma unroll
    for (int nt = 0; nt < 8; ++nt) {
      v16h b = frag_b(&We2h[(nt * 16 + nl) * 128 + kt * 32 + 8 * hh]);
      acc[nt] = WMMA(a, b, acc[nt]);
    }
  }
  __syncthreads();  // all m1 reads done before overwriting M1 with m
#pragma unroll
  for (int nt = 0; nt < 8; ++nt) {
    int n = nt * 16 + nl;
#pragma unroll
    for (int j = 0; j < 8; ++j) {
      float v = silu_f(acc[nt][j]);
      int M = j + 8 * hh;
      M1[M * 128 + n] = (_Float16)v;
      atomicAdd(&agg[(long long)dstS[M] * HDIM + n], v);   // segment_sum(m, dst)
    }
  }
  if (lane < 16) atomicAdd(&cnt[dstS[lane]], 1.0f);
  __syncthreads();

  // --- GEMM3: cw = silu(m @ Wc1.T + bc1) @ Wc2.T, K=128 --------------------
#pragma unroll
  for (int nt = 0; nt < 8; ++nt) {
    float b0 = bc1[nt * 16 + nl];
#pragma unroll
    for (int j = 0; j < 8; ++j) acc[nt][j] = b0;
  }
#pragma unroll
  for (int kt = 0; kt < 4; ++kt) {
    v16h a = frag_a(&M1[nl * 128 + kt * 32 + 8 * hh]);
#pragma unroll
    for (int nt = 0; nt < 8; ++nt) {
      v16h b = frag_b(&Wc1h[(nt * 16 + nl) * 128 + kt * 32 + 8 * hh]);
      acc[nt] = WMMA(a, b, acc[nt]);
    }
  }
#pragma unroll
  for (int nt = 0; nt < 8; ++nt) {
    float w2 = Wc2[nt * 16 + nl];
#pragma unroll
    for (int j = 0; j < 8; ++j)
      atomicAdd(&cwS[j + 8 * hh], silu_f(acc[nt][j]) * w2);
  }
  __syncthreads();

  if (lane < 16) {
    float cw = cwS[lane];
    int d = dstS[lane];
    atomicAdd(&coordAgg[d * 3 + 0], diffS[lane * 3 + 0] * cw);
    atomicAdd(&coordAgg[d * 3 + 1], diffS[lane * 3 + 1] * cw);
    atomicAdd(&coordAgg[d * 3 + 2], diffS[lane * 3 + 2] * cw);
  }
}

// ---------------------------------------------------------------------------
// 3) node MLP + residual + layernorm: one wave per 16 nodes.
__global__ __launch_bounds__(32) void egnn_node_kernel(
    const float* __restrict__ h, const float* __restrict__ agg,
    const float* __restrict__ cnt,
    const float* __restrict__ bn1, const float* __restrict__ bn2,
    const _Float16* __restrict__ Wn1h, const _Float16* __restrict__ Wn2h,
    const float* __restrict__ gamma, const float* __restrict__ beta,
    float* __restrict__ hout) {
  __shared__ _Float16 A[16 * 256];   // [node][h | agg/cnt]
  __shared__ _Float16 U[16 * 128];
  __shared__ float Y[16 * 128];
  __shared__ float rsum[16], rsq[16], invc[16];

  const int lane = threadIdx.x;
  const int n0 = blockIdx.x * 16;
  const int nl = lane & 15;
  const int hh = lane >> 4;

  if (lane < 16) {
    invc[lane] = 1.f / fmaxf(cnt[n0 + lane], 1.f);
    rsum[lane] = 0.f; rsq[lane] = 0.f;
  }
  __syncthreads();

  for (int r = 0; r < 16; ++r) {
    long long row = n0 + r;
    float4 vh = *reinterpret_cast<const float4*>(h + row * HDIM + lane * 4);
    float4 va = *reinterpret_cast<const float4*>(agg + row * HDIM + lane * 4);
    float ic = invc[r];
    _Float16* dp = &A[r * 256 + lane * 4];
    dp[0] = (_Float16)vh.x; dp[1] = (_Float16)vh.y;
    dp[2] = (_Float16)vh.z; dp[3] = (_Float16)vh.w;
    _Float16* da = &A[r * 256 + 128 + lane * 4];
    da[0] = (_Float16)(va.x * ic); da[1] = (_Float16)(va.y * ic);
    da[2] = (_Float16)(va.z * ic); da[3] = (_Float16)(va.w * ic);
  }
  __syncthreads();

  v8f acc[8];

  // u = silu([h|agg] @ Wn1.T + bn1), K=256
#pragma unroll
  for (int nt = 0; nt < 8; ++nt) {
    float b0 = bn1[nt * 16 + nl];
#pragma unroll
    for (int j = 0; j < 8; ++j) acc[nt][j] = b0;
  }
#pragma unroll
  for (int kt = 0; kt < 8; ++kt) {
    v16h a = frag_a(&A[nl * 256 + kt * 32 + 8 * hh]);
#pragma unroll
    for (int nt = 0; nt < 8; ++nt) {
      v16h b = frag_b(&Wn1h[(nt * 16 + nl) * 256 + kt * 32 + 8 * hh]);
      acc[nt] = WMMA(a, b, acc[nt]);
    }
  }
#pragma unroll
  for (int nt = 0; nt < 8; ++nt) {
    int n = nt * 16 + nl;
#pragma unroll
    for (int j = 0; j < 8; ++j)
      U[(j + 8 * hh) * 128 + n] = (_Float16)silu_f(acc[nt][j]);
  }
  __syncthreads();

  // y = h + u @ Wn2.T + bn2, K=128
#pragma unroll
  for (int nt = 0; nt < 8; ++nt) {
    float b0 = bn2[nt * 16 + nl];
#pragma unroll
    for (int j = 0; j < 8; ++j) acc[nt][j] = b0;
  }
#pragma unroll
  for (int kt = 0; kt < 4; ++kt) {
    v16h a = frag_a(&U[nl * 128 + kt * 32 + 8 * hh]);
#pragma unroll
    for (int nt = 0; nt < 8; ++nt) {
      v16h b = frag_b(&Wn2h[(nt * 16 + nl) * 128 + kt * 32 + 8 * hh]);
      acc[nt] = WMMA(a, b, acc[nt]);
    }
  }
#pragma unroll
  for (int nt = 0; nt < 8; ++nt) {
    int n = nt * 16 + nl;
#pragma unroll
    for (int j = 0; j < 8; ++j) {
      int M = j + 8 * hh;
      float y = acc[nt][j] + h[(long long)(n0 + M) * HDIM + n];  // f32 residual
      Y[M * 128 + n] = y;
      atomicAdd(&rsum[M], y);
      atomicAdd(&rsq[M], y * y);
    }
  }
  __syncthreads();

  // layernorm
#pragma unroll
  for (int nt = 0; nt < 8; ++nt) {
    int n = nt * 16 + nl;
    float g = gamma[n], bb = beta[n];
#pragma unroll
    for (int j = 0; j < 8; ++j) {
      int M = j + 8 * hh;
      float mu = rsum[M] * (1.f / 128.f);
      float var = rsq[M] * (1.f / 128.f) - mu * mu;
      float y = Y[M * 128 + n];
      hout[(long long)(n0 + M) * HDIM + n] = (y - mu) * rsqrtf(var + 1e-5f) * g + bb;
    }
  }
}

// ---------------------------------------------------------------------------
// 4) x_out = x + coord_agg / max(cnt,1)
__global__ void egnn_xout_kernel(const float* __restrict__ x,
                                 const float* __restrict__ coordAgg,
                                 const float* __restrict__ cnt,
                                 float* __restrict__ outx) {
  int i = blockIdx.x * 256 + threadIdx.x;
  if (i < N_NODES * 3) {
    float c = fmaxf(cnt[i / 3], 1.f);
    outx[i] = x[i] + coordAgg[i] / c;
  }
}

// ---------------------------------------------------------------------------
extern "C" void kernel_launch(void* const* d_in, const int* in_sizes, int n_in,
                              void* d_out, int out_size, void* d_ws, size_t ws_size,
                              hipStream_t stream) {
  const float*     h     = (const float*)d_in[0];
  const float*     x     = (const float*)d_in[1];
  const long long* ei    = (const long long*)d_in[2];   // int64 edge_index (2,E)
  const float*     We1   = (const float*)d_in[3];
  const float*     be1   = (const float*)d_in[4];
  const float*     We2   = (const float*)d_in[5];
  const float*     be2   = (const float*)d_in[6];
  const float*     Wn1   = (const float*)d_in[7];
  const float*     bn1   = (const float*)d_in[8];
  const float*     Wn2   = (const float*)d_in[9];
  const float*     bn2   = (const float*)d_in[10];
  const float*     Wc1   = (const float*)d_in[11];
  const float*     bc1   = (const float*)d_in[12];
  const float*     Wc2   = (const float*)d_in[13];
  const float*     gamma = (const float*)d_in[14];
  const float*     beta  = (const float*)d_in[15];

  float* hout = (float*)d_out;                       // (N,128)
  float* xout = hout + (size_t)N_NODES * HDIM;       // (N,3)

  // workspace layout
  float* agg      = (float*)d_ws;                    // N*128
  float* cnt      = agg + (size_t)N_NODES * HDIM;    // N
  float* coordAgg = cnt + N_NODES;                   // N*3
  _Float16* We1h  = (_Float16*)(coordAgg + (size_t)N_NODES * 3);
  _Float16* We2h  = We1h + 128 * 256;
  _Float16* Wn1h  = We2h + 128 * 128;
  _Float16* Wn2h  = Wn1h + 128 * 256;
  _Float16* Wc1h  = Wn2h + 128 * 128;

  long long nzero = (long long)N_NODES * (HDIM + 1 + 3);   // agg+cnt+coordAgg contiguous
  egnn_zero_kernel<<<(int)((nzero + 255) / 256), 256, 0, stream>>>(agg, nzero);
  egnn_cvt_kernel<<<128, 256, 0, stream>>>(We1, We2, Wn1, Wn2, Wc1,
                                           We1h, We2h, Wn1h, Wn2h, Wc1h);
  egnn_edge_kernel<<<N_EDGES / 16, 32, 0, stream>>>(h, x, ei, We1, be1, be2, bc1, Wc2,
                                                    We1h, We2h, Wc1h,
                                                    agg, cnt, coordAgg);
  egnn_node_kernel<<<N_NODES / 16, 32, 0, stream>>>(h, agg, cnt, bn1, bn2,
                                                    Wn1h, Wn2h, gamma, beta, hout);
  egnn_xout_kernel<<<(N_NODES * 3 + 255) / 256, 256, 0, stream>>>(x, coordAgg, cnt, xout);
}
